// GraphEncoder_76630806495728
// MI455X (gfx1250) — compile-verified, matrix operand order checked by reading, not run on hardware
//
#include <hip/hip_runtime.h>

typedef __bf16 bf16_t;
typedef __attribute__((ext_vector_type(4)))  bf16_t v4bf;
typedef __attribute__((ext_vector_type(8)))  bf16_t v8bf;
typedef __attribute__((ext_vector_type(16))) bf16_t v16bf;
typedef __attribute__((ext_vector_type(4)))  float  v4f;
typedef __attribute__((ext_vector_type(8)))  float  v8f;
typedef __attribute__((ext_vector_type(4)))  unsigned int u32x4;
typedef __attribute__((ext_vector_type(4)))  int    i32x4;
typedef __attribute__((ext_vector_type(8)))  int    i32x8;

#define NN    4096
#define INC   256
#define HIDC  512
#define OUTC  256
#define BATCH 8
#define BT    128   // block tile (M and N)
#define KT    32    // K step per WMMA pass
#define LSTR  40    // padded LDS row stride in bf16 elems (80B, conflict-free)

#if defined(__has_builtin)
#if __has_builtin(__builtin_amdgcn_tensor_load_to_lds)
#define HAVE_TDM 1
#endif
#endif

// ---------------- WMMA helpers ----------------
__device__ __forceinline__ v8f wmma_bf16(v16bf a, v16bf b, v8f c) {
  return __builtin_amdgcn_wmma_f32_16x16x32_bf16(
      /*neg_a=*/false, a, /*neg_b=*/false, b,
      /*c_mod=*/(short)0, c, /*reuse_a=*/false, /*reuse_b=*/false);
}

// Load one 16x32 bf16 operand fragment from an LDS tile with row stride LSTR.
__device__ __forceinline__ v16bf load_frag(const bf16_t* tile, int base_row, int lane) {
  int row   = base_row + (lane & 15);
  int half8 = (lane >> 4) * 8;
  const bf16_t* p = tile + row * LSTR + half8;
  union { v16bf v; v8bf h[2]; } u;
  u.h[0] = *(const v8bf*)(p);        // K 0..7   (or 8..15)
  u.h[1] = *(const v8bf*)(p + 16);   // K 16..23 (or 24..31)
  return u.v;
}

// ---------------- K0: degree -> d^{-1/2} ----------------
__global__ void deg_kernel(const float* __restrict__ A, float* __restrict__ dis) {
  __shared__ float red[256];
  int row = blockIdx.x;                       // 0 .. B*N-1
  const float* a = A + (size_t)row * NN;
  float s = 0.f;
  for (int c = threadIdx.x; c < NN; c += 256) s += a[c];
  red[threadIdx.x] = s;
  __syncthreads();
  for (int off = 128; off > 0; off >>= 1) {
    if (threadIdx.x < off) red[threadIdx.x] += red[threadIdx.x + off];
    __syncthreads();
  }
  if (threadIdx.x == 0) {
    float deg = red[0] + 1.0f;                // + self loop
    dis[row] = (deg > 0.f) ? rsqrtf(fmaxf(deg, 1e-12f)) : 0.f;
  }
}

// ---------------- K1: transpose+bf16 weights / bf16 convert x ----------------
__global__ void wt_kernel(const float* __restrict__ W, bf16_t* __restrict__ Wt,
                          int C, int F) {          // W[C][F] -> Wt[F][C]
  int idx = blockIdx.x * 256 + threadIdx.x;
  if (idx < C * F) {
    int c = idx / F, f = idx % F;
    Wt[(size_t)f * C + c] = (bf16_t)W[idx];
  }
}

__global__ void cvt_kernel(const float* __restrict__ X, bf16_t* __restrict__ Xb, int n) {
  int i = blockIdx.x * 256 + threadIdx.x;
  if (i < n) Xb[i] = (bf16_t)X[i];
}

// ---------------- Weight GEMM: Yt[b][f][j] = dis[b][j] * sum_c Wt[f][c]*Xb[b][j][c] ----
template <int K_TOTAL, int M_TOTAL>
__global__ void __launch_bounds__(256)
wgemm_kernel(const bf16_t* __restrict__ Wt,   // [M_TOTAL][K_TOTAL]
             const bf16_t* __restrict__ Xb,   // [B][NN][K_TOTAL]
             const float*  __restrict__ dis,  // [B][NN]
             bf16_t* __restrict__ Yt) {       // [B][M_TOTAL][NN]
  int b = blockIdx.z;
  int nBase = blockIdx.x * BT;                // node (N-dim) base
  int mBase = blockIdx.y * BT;                // feature (M-dim) base
  const bf16_t* X = Xb + (size_t)b * NN * K_TOTAL;
  const float*  d = dis + (size_t)b * NN;
  bf16_t*       Y = Yt + (size_t)b * M_TOTAL * NN;

  __shared__ bf16_t As[BT * LSTR];
  __shared__ bf16_t Bs[BT * LSTR];

  int t = threadIdx.x, lane = t & 31, wave = t >> 5;
  int wm = wave & 3, wn = wave >> 2;          // 4x2 wave grid

  v8f acc[2][4];
#pragma unroll
  for (int i = 0; i < 2; ++i)
#pragma unroll
    for (int j = 0; j < 4; ++j) acc[i][j] = (v8f)0.0f;

  for (int k0 = 0; k0 < K_TOTAL; k0 += KT) {
    __syncthreads();
#pragma unroll
    for (int it = 0; it < 2; ++it) {          // fill A (Wt rows) and B (X rows)
      int r  = (t >> 2) + it * 64;
      int c8 = (t & 3) * 8;
      *(v8bf*)&As[r * LSTR + c8] = *(const v8bf*)&Wt[(size_t)(mBase + r) * K_TOTAL + k0 + c8];
      *(v8bf*)&Bs[r * LSTR + c8] = *(const v8bf*)&X[(size_t)(nBase + r) * K_TOTAL + k0 + c8];
    }
    __syncthreads();

    v16bf af[2], bf[4];
#pragma unroll
    for (int tm = 0; tm < 2; ++tm) af[tm] = load_frag(As, wm * 32 + tm * 16, lane);
#pragma unroll
    for (int tn = 0; tn < 4; ++tn) bf[tn] = load_frag(Bs, wn * 64 + tn * 16, lane);
#pragma unroll
    for (int tm = 0; tm < 2; ++tm)
#pragma unroll
      for (int tn = 0; tn < 4; ++tn) acc[tm][tn] = wmma_bf16(af[tm], bf[tn], acc[tm][tn]);
  }

  // epilogue: scale by dis[j], store transposed bf16
  int mo = (lane >> 4) * 8;
#pragma unroll
  for (int tm = 0; tm < 2; ++tm)
#pragma unroll
    for (int tn = 0; tn < 4; ++tn) {
      int col = nBase + wn * 64 + tn * 16 + (lane & 15);   // node j
      int i0  = mBase + wm * 32 + tm * 16 + mo;            // feature f
      float dv = d[col];
#pragma unroll
      for (int r = 0; r < 8; ++r)
        Y[(size_t)(i0 + r) * NN + col] = (bf16_t)(acc[tm][tn][r] * dv);
    }
}

// ---------------- A-GEMM: out[b][i][f] = dis[i]*(sum_j A[i][j]*Yt[f][j] + Yt[f][i]) + bias[f]
template <int F_TOTAL, bool RELU_BF16>
__global__ void __launch_bounds__(256)
agemm_kernel(const float*  __restrict__ A,    // [B][NN][NN] f32 (converted on the fly)
             const bf16_t* __restrict__ Yt,   // [B][F_TOTAL][NN] bf16
             const float*  __restrict__ dis,  // [B][NN]
             const float*  __restrict__ bias, // [F_TOTAL]
             bf16_t* __restrict__ Hout,       // RELU_BF16: [B][NN][F_TOTAL]
             float*  __restrict__ Fout) {     // else:      [B][NN][F_TOTAL]
  int b = blockIdx.z;
  int nBase = blockIdx.x * BT;                // feature base (N-dim)
  int mBase = blockIdx.y * BT;                // node-row base (M-dim)
  const float*  Ab = A  + (size_t)b * NN * NN;
  const bf16_t* Y  = Yt + (size_t)b * F_TOTAL * NN;
  const float*  d  = dis + (size_t)b * NN;

  __shared__ bf16_t As[BT * LSTR];
  __shared__ bf16_t Bs[BT * LSTR];

  int t = threadIdx.x, lane = t & 31, wave = t >> 5;
  int wm = wave & 3, wn = wave >> 2;

  v8f acc[2][4];
#pragma unroll
  for (int i = 0; i < 2; ++i)
#pragma unroll
    for (int j = 0; j < 4; ++j) acc[i][j] = (v8f)0.0f;

#ifdef HAVE_TDM
  // --- Tensor Data Mover descriptor (static parts) for the B tile:
  // 2D tile: tile_dim0 = 32 bf16 (64B rows), tile_dim1 = 128 rows,
  // tensor_dim0_stride = NN; LDS pad: every 16 DWORDs insert 4 DWORDs
  // -> reproduces the 80B (LSTR) conflict-free LDS row stride.
  unsigned ldsB = (unsigned)(unsigned long long)(const void*)Bs;  // LDS byte offset
  unsigned long long gpanel = (unsigned long long)(const void*)(Y + (size_t)nBase * NN);
  i32x8 g1;
  g1[0] = (1 << 16)        // data_size = 2 bytes
        | (1 << 20)        // pad_enable
        | (3 << 22)        // pad_interval: 16 DWORDs (64B) per row
        | (3 << 25);       // pad_amount:   4 DWORDs (16B)
  g1[1] = (int)((NN & 0xFFFF) << 16);          // tensor_dim0[15:0]
  g1[2] = (int)((unsigned)F_TOTAL << 16);      // tensor_dim1[15:0]
  g1[3] = (int)(32u << 16);                    // tile_dim0 = 32
  g1[4] = 128;                                 // tile_dim1 = 128
  g1[5] = NN;                                  // tensor_dim0_stride[31:0]
  g1[6] = 0;
  g1[7] = 0;
#endif

  for (int k0 = 0; k0 < NN; k0 += KT) {
    __syncthreads();

#ifdef HAVE_TDM
    // B tile via TDM (wave 0 issues; DMA overlaps the A-tile f32->bf16 work)
    if (t < 32) {
      unsigned long long ga = gpanel + (unsigned long long)k0 * 2u;
      u32x4 g0;
      g0[0] = 1u;                                            // count=1
      g0[1] = ldsB;                                          // lds_addr
      g0[2] = (unsigned)(ga & 0xFFFFFFFFull);                // global_addr lo
      g0[3] = (unsigned)((ga >> 32) & 0x1FFFFFFull) | (2u << 30);  // hi | type=2
      __builtin_amdgcn_tensor_load_to_lds(g0, g1, (i32x4)0, (i32x4)0, (i32x8)0, 0);
    }
#endif

    // A tile: 128 rows x 32 f32, convert f32 -> bf16 in registers, ds_store_b64
#pragma unroll
    for (int it = 0; it < 4; ++it) {
      int r  = (t >> 3) + it * 32;
      int c4 = (t & 7) * 4;
      v4f av = *(const v4f*)&Ab[(size_t)(mBase + r) * NN + k0 + c4];
      v4bf bv;
      bv[0] = (bf16_t)av[0]; bv[1] = (bf16_t)av[1];
      bv[2] = (bf16_t)av[2]; bv[3] = (bf16_t)av[3];
      *(v4bf*)&As[r * LSTR + c4] = bv;
      if (k0 + KT < NN)  // prefetch next A k-tile
        __builtin_prefetch(&Ab[(size_t)(mBase + r) * NN + k0 + KT + c4], 0, 1);
    }

#ifdef HAVE_TDM
    if (t < 32) __builtin_amdgcn_s_wait_tensorcnt(0);        // B tile landed in LDS
#else
#pragma unroll
    for (int it = 0; it < 2; ++it) {
      int r  = (t >> 2) + it * 64;
      int c8 = (t & 3) * 8;
      *(v8bf*)&Bs[r * LSTR + c8] = *(const v8bf*)&Y[(size_t)(nBase + r) * NN + k0 + c8];
    }
#endif
    __syncthreads();

    v16bf af[2], bf[4];
#pragma unroll
    for (int tm = 0; tm < 2; ++tm) af[tm] = load_frag(As, wm * 32 + tm * 16, lane);
#pragma unroll
    for (int tn = 0; tn < 4; ++tn) bf[tn] = load_frag(Bs, wn * 64 + tn * 16, lane);
#pragma unroll
    for (int tm = 0; tm < 2; ++tm)
#pragma unroll
      for (int tn = 0; tn < 4; ++tn) acc[tm][tn] = wmma_bf16(af[tm], bf[tn], acc[tm][tn]);
  }

  // epilogue: + self-loop term, * dis[i], + bias, (relu), store
  int mo = (lane >> 4) * 8;
#pragma unroll
  for (int tm = 0; tm < 2; ++tm)
#pragma unroll
    for (int tn = 0; tn < 4; ++tn) {
      int col = nBase + wn * 64 + tn * 16 + (lane & 15);   // feature f
      int i0  = mBase + wm * 32 + tm * 16 + mo;            // node i
      float bv = bias[col];
#pragma unroll
      for (int r = 0; r < 8; ++r) {
        int i = i0 + r;
        float self = (float)Y[(size_t)col * NN + i];       // Yt[f][i] (I term)
        float v = d[i] * (acc[tm][tn][r] + self) + bv;
        if (RELU_BF16) {
          v = fmaxf(v, 0.0f);
          Hout[((size_t)b * NN + i) * F_TOTAL + col] = (bf16_t)v;
        } else {
          Fout[((size_t)b * NN + i) * F_TOTAL + col] = v;
        }
      }
    }
}

// ---------------- host launcher ----------------
extern "C" void kernel_launch(void* const* d_in, const int* in_sizes, int n_in,
                              void* d_out, int out_size, void* d_ws, size_t ws_size,
                              hipStream_t stream) {
  const float* x  = (const float*)d_in[0];
  const float* A  = (const float*)d_in[1];
  const float* W1 = (const float*)d_in[2];
  const float* b1 = (const float*)d_in[3];
  const float* W2 = (const float*)d_in[4];
  const float* b2 = (const float*)d_in[5];
  float* out = (float*)d_out;

  char* p = (char*)d_ws;
  float*  dis = (float*)p;  p += (size_t)BATCH * NN * sizeof(float);
  bf16_t* W1t = (bf16_t*)p; p += (size_t)HIDC * INC * sizeof(bf16_t);
  bf16_t* W2t = (bf16_t*)p; p += (size_t)OUTC * HIDC * sizeof(bf16_t);
  bf16_t* xbf = (bf16_t*)p; p += (size_t)BATCH * NN * INC * sizeof(bf16_t);
  bf16_t* Y1t = (bf16_t*)p; p += (size_t)BATCH * HIDC * NN * sizeof(bf16_t);
  bf16_t* Hbf = (bf16_t*)p; p += (size_t)BATCH * NN * HIDC * sizeof(bf16_t);
  bf16_t* Y2t = (bf16_t*)p; p += (size_t)BATCH * OUTC * NN * sizeof(bf16_t);

  // 1) dis = (rowsum(A)+1)^{-1/2}
  deg_kernel<<<BATCH * NN, 256, 0, stream>>>(A, dis);
  // 2) precision/layout prep
  wt_kernel<<<(INC * HIDC + 255) / 256, 256, 0, stream>>>(W1, W1t, INC, HIDC);
  wt_kernel<<<(HIDC * OUTC + 255) / 256, 256, 0, stream>>>(W2, W2t, HIDC, OUTC);
  cvt_kernel<<<((size_t)BATCH * NN * INC + 255) / 256, 256, 0, stream>>>(x, xbf, BATCH * NN * INC);
  // 3) Y1t = dis ⊙ (x@W1)  (transposed bf16)
  wgemm_kernel<INC, HIDC><<<dim3(NN / BT, HIDC / BT, BATCH), 256, 0, stream>>>(W1t, xbf, dis, Y1t);
  // 4) H = relu(dis ⊙ (A@Y1 + Y1) + b1)
  agemm_kernel<HIDC, true><<<dim3(HIDC / BT, NN / BT, BATCH), 256, 0, stream>>>(
      A, Y1t, dis, b1, Hbf, nullptr);
  // 5) Y2t = dis ⊙ (H@W2)
  wgemm_kernel<HIDC, OUTC><<<dim3(NN / BT, OUTC / BT, BATCH), 256, 0, stream>>>(W2t, Hbf, dis, Y2t);
  // 6) out = dis ⊙ (A@Y2 + Y2) + b2
  agemm_kernel<OUTC, false><<<dim3(OUTC / BT, NN / BT, BATCH), 256, 0, stream>>>(
      A, Y2t, dis, b2, nullptr, out);
}
